// DifferentialEntropyRegularization_18124761989505
// MI455X (gfx1250) — matrix operand order repaired and checked by reading, not compiled, not run on hardware
//
#include <hip/hip_runtime.h>
#include <hip/hip_bf16.h>

// Problem constants (from reference): x is [16384, 512] fp32, scalar output.
#define N   16384
#define D   512
#define NT  (N / 16)      // 1024 column tiles of 16
#define ROWS_PER_WG 64    // 4 strips of 16 rows per wave (register B reuse)
#define WEIGHT    0.1f
#define EPS       1e-8f
#define PDIST_EPS 1e-6f

typedef __attribute__((ext_vector_type(16))) __bf16 v16bf;
typedef __attribute__((ext_vector_type(4)))  __bf16 v4bf;
typedef __attribute__((ext_vector_type(8)))  float  v8f;

// ---------------------------------------------------------------------------
// Kernel 0: zero the scalar output (harness poisons d_out with 0xAA).
// ---------------------------------------------------------------------------
__global__ void zero_out_kernel(float* out) {
    if (threadIdx.x == 0 && blockIdx.x == 0) out[0] = 0.0f;
}

// ---------------------------------------------------------------------------
// Kernel 1: fp32 -> bf16 hi/lo split.  hi = bf16(x), lo = bf16(x - hi).
// hi*hi + hi*lo + lo*hi recovers ~fp32 accuracy on the WMMA path.
// ---------------------------------------------------------------------------
__global__ void cvt_split_kernel(const float* __restrict__ x,
                                 __bf16* __restrict__ hi,
                                 __bf16* __restrict__ lo,
                                 int n4) {
    int t = blockIdx.x * blockDim.x + threadIdx.x;
    if (t >= n4) return;
    float4 v = ((const float4*)x)[t];
    float f[4] = {v.x, v.y, v.z, v.w};
    v4bf h, l;
#pragma unroll
    for (int k = 0; k < 4; ++k) {
        __bf16 hk = (__bf16)f[k];
        h[k] = hk;
        l[k] = (__bf16)(f[k] - (float)hk);
    }
    ((v4bf*)hi)[t] = h;
    ((v4bf*)lo)[t] = l;
}

// Pack (value, index) into a 64-bit key whose unsigned max == argmax.
// Index stored inverted so exact ties pick the SMALLER index (jnp.argmax).
__device__ __forceinline__ unsigned long long pack_key(float v, int idx) {
    unsigned fb = __float_as_uint(v);
    fb = (fb & 0x80000000u) ? ~fb : (fb | 0x80000000u);
    return ((unsigned long long)fb << 32) | (unsigned)(~idx);
}

// ---------------------------------------------------------------------------
// Kernel 2: fused Gram-matrix (x @ x^T) + row argmax, diagonal set to -1.0.
// Workgroup (8 wave32s) owns 64 A rows.  The 128 KB hi+lo A-tile is staged
// by async global->LDS DMA (ASYNCcnt) *pre-swizzled into WMMA fragment
// order*: lane L's 32-byte A fragment for (k-step kk, strip s, array a) sits
// contiguously at  kk*8192 + (2s+a)*1024 + L*32 , so the hot loop addresses
// all 8 fragments per k-step with small immediate DS offsets (no per-step
// VALU address math -> no WMMA->VALU hazard NOPs).  Each wave computes a
// 64x16 C strip per column tile: one B fragment pair (global, L2-resident)
// feeds 4 A strips -> 12 WMMAs per 2 B loads (4x less L2 B-traffic).
// ---------------------------------------------------------------------------
__global__ void __launch_bounds__(256)
gram_argmax_kernel(const __bf16* __restrict__ hi,
                   const __bf16* __restrict__ lo,
                   int* __restrict__ nn) {
    extern __shared__ __bf16 smem[];           // 128 KB fragment-ordered A
    __shared__ unsigned long long rowbest[ROWS_PER_WG];

    const int i0   = blockIdx.x * ROWS_PER_WG; // this block's 64 rows
    const int wave = threadIdx.x >> 5;         // 0..7
    const int lane = threadIdx.x & 31;         // wave32
    const int nIdx = lane & 15;                // N (col) / M (row-in-strip)
    const int hgrp = lane >> 4;                // lane half-group (K split)

    // ---- Stage A-tile (64 rows, hi+lo) swizzled via async DMA ----
    // 8192 16-byte chunks; chunk id -> (kk, s, a, L, c):
    //   dest  = id*16  =  (((kk*8 + s*2 + a)*32 + L)*2 + c) * 16
    //   src   = row (s*16 + L%16), elems kk*32 + (L/16)*16 + c*8 .. +7
    {
        const unsigned long long ghi =
            (unsigned long long)(size_t)(hi + (size_t)i0 * D);
        const unsigned long long glo =
            (unsigned long long)(size_t)(lo + (size_t)i0 * D);
        const unsigned ldsbase = (unsigned)(size_t)(void*)smem;
#pragma unroll
        for (int p = 0; p < 32; ++p) {
            const int id = threadIdx.x + 256 * p;
            const int c  = id & 1;
            const int L  = (id >> 1) & 31;
            const int a  = (id >> 6) & 1;
            const int s  = (id >> 7) & 3;
            const int kk = id >> 9;
            const int m = L & 15, h = L >> 4;
            const int src_elem = (s * 16 + m) * D + kk * 32 + h * 16 + c * 8;
            unsigned long long ga =
                (a ? glo : ghi) + (unsigned long long)(src_elem * 2);
            unsigned la = ldsbase + (unsigned)(id * 16);
            asm volatile("global_load_async_to_lds_b128 %0, %1, off"
                         :: "v"(la), "v"(ga) : "memory");
        }
    }
    if (threadIdx.x < ROWS_PER_WG) rowbest[threadIdx.x] = 0ull;
    asm volatile("s_wait_asynccnt 0x0" ::: "memory");
    __syncthreads();

    // Per-lane running argmax: strip s, VGPR r -> row i0 + 16s + r + 8*hgrp.
    float best[4][8];
    int   bidx[4][8];
#pragma unroll
    for (int s = 0; s < 4; ++s)
#pragma unroll
        for (int r = 0; r < 8; ++r) { best[s][r] = -__builtin_inff(); bidx[s][r] = 0; }

    // Each wave handles column tiles wave, wave+8, ...
    for (int jt = wave; jt < NT; jt += 8) {
        const int j0 = jt * 16;
        // B-matrix (K x N): lane covers column j0+nIdx; contiguous K run.
        const __bf16* bh = hi + (size_t)(j0 + nIdx) * D + hgrp * 16;
        const __bf16* bl = lo + (size_t)(j0 + nIdx) * D + hgrp * 16;

        v8f c0[4], c1[4], c2[4];
#pragma unroll
        for (int s = 0; s < 4; ++s) { c0[s] = (v8f){}; c1[s] = (v8f){}; c2[s] = (v8f){}; }

#pragma unroll 2
        for (int kk = 0; kk < 16; ++kk) {
            // One B fragment pair feeds all 4 A strips (12 WMMAs / 2 loads).
            v16bf bhi = *(const v16bf*)&bh[kk * 32];
            v16bf blo = *(const v16bf*)&bl[kk * 32];
            // Fragment-ordered A: per-lane base + small immediate offsets.
            const int fb = kk * 4096 + lane * 16;   // elements
#pragma unroll
            for (int s = 0; s < 4; ++s) {
                v16bf ahi = *(const v16bf*)&smem[fb + (s * 2 + 0) * 512];
                v16bf alo = *(const v16bf*)&smem[fb + (s * 2 + 1) * 512];
                c0[s] = __builtin_amdgcn_wmma_f32_16x16x32_bf16(
                            false, ahi, false, bhi, (short)0, c0[s], false, false);
                c1[s] = __builtin_amdgcn_wmma_f32_16x16x32_bf16(
                            false, ahi, false, blo, (short)0, c1[s], false, false);
                c2[s] = __builtin_amdgcn_wmma_f32_16x16x32_bf16(
                            false, alo, false, bhi, (short)0, c2[s], false, false);
            }
        }

        const int gcol = j0 + nIdx;
#pragma unroll
        for (int s = 0; s < 4; ++s) {
            v8f c = c0[s] + (c1[s] + c2[s]);
#pragma unroll
            for (int r = 0; r < 8; ++r) {
                const int grow = i0 + s * 16 + r + hgrp * 8;
                float v = c[r];
                if (grow == gcol) v = -1.0f;      // mask self-similarity
                if (v > best[s][r]) { best[s][r] = v; bidx[s][r] = gcol; }
            }
        }
    }

    // Merge lanes/waves per row via packed u64 max (ds_max_u64).
#pragma unroll
    for (int s = 0; s < 4; ++s)
#pragma unroll
        for (int r = 0; r < 8; ++r)
            atomicMax(&rowbest[s * 16 + r + hgrp * 8],
                      pack_key(best[s][r], bidx[s][r]));
    __syncthreads();
    if (threadIdx.x < ROWS_PER_WG)
        nn[i0 + threadIdx.x] =
            (int)~(unsigned)(rowbest[threadIdx.x] & 0xFFFFFFFFull);
}

// ---------------------------------------------------------------------------
// Kernel 3: loss = -WEIGHT * mean(log(||x_i - x_nn(i) + eps_d|| + eps)).
// One wave32 per row; coalesced float4 loads; shfl reduction; scaled
// atomic add into the scalar output.
// ---------------------------------------------------------------------------
__global__ void __launch_bounds__(256)
loss_kernel(const float* __restrict__ x,
            const int* __restrict__ nn,
            float* __restrict__ out) {
    const int row  = (blockIdx.x * blockDim.x + threadIdx.x) >> 5;
    const int lane = threadIdx.x & 31;
    if (row >= N) return;
    const int j = nn[row];
    const float4* a4 = (const float4*)(x + (size_t)row * D);
    const float4* b4 = (const float4*)(x + (size_t)j   * D);
    float s = 0.0f;
#pragma unroll
    for (int c = 0; c < 4; ++c) {
        float4 av = a4[lane + 32 * c];
        float4 bv = b4[lane + 32 * c];
        float d0 = av.x - bv.x + PDIST_EPS;
        float d1 = av.y - bv.y + PDIST_EPS;
        float d2 = av.z - bv.z + PDIST_EPS;
        float d3 = av.w - bv.w + PDIST_EPS;
        s += d0 * d0 + d1 * d1 + d2 * d2 + d3 * d3;
    }
#pragma unroll
    for (int m = 16; m > 0; m >>= 1) s += __shfl_xor(s, m, 32);
    if (lane == 0) {
        float rho = sqrtf(s);
        atomicAdd(out, -logf(rho + EPS) * (WEIGHT / (float)N));
    }
}

// ---------------------------------------------------------------------------
extern "C" void kernel_launch(void* const* d_in, const int* in_sizes, int n_in,
                              void* d_out, int out_size, void* d_ws, size_t ws_size,
                              hipStream_t stream) {
    (void)in_sizes; (void)n_in; (void)out_size; (void)ws_size;
    const float* x = (const float*)d_in[0];
    float* out = (float*)d_out;

    // Workspace layout: bf16 hi (16 MB) | bf16 lo (16 MB) | nn idx (64 KB)
    __bf16* hi = (__bf16*)d_ws;
    __bf16* lo = hi + (size_t)N * D;
    int*    nn = (int*)(lo + (size_t)N * D);

    zero_out_kernel<<<1, 1, 0, stream>>>(out);

    const int n4 = (N * D) / 4;
    cvt_split_kernel<<<(n4 + 255) / 256, 256, 0, stream>>>(x, hi, lo, n4);

    // 128 KB dynamic LDS for the fragment-ordered 64-row hi+lo A tile
    // (CDNA5: 320 KB LDS per WGP).
    const size_t lds_bytes = (size_t)2 * ROWS_PER_WG * D * sizeof(__bf16);
    gram_argmax_kernel<<<N / ROWS_PER_WG, 256, lds_bytes, stream>>>(hi, lo, nn);

    loss_kernel<<<(N * 32) / 256, 256, 0, stream>>>(x, nn, out);
}
